// DualCrossModalAttention_89094801588667
// MI455X (gfx1250) — matrix operand order repaired and checked by reading, not compiled, not run on hardware
//
#include <hip/hip_runtime.h>

// ---------------------------------------------------------------------------
// DualCrossModalAttention on MI455X (gfx1250).
// All GEMMs via v_wmma_f32_16x16x32_bf16 (wave32), f32 accumulate.
// Global->LDS staging uses CDNA5 async copies (GLOBAL_LOAD_ASYNC_TO_LDS_B128,
// ASYNCcnt-tracked, no VGPR transit), double-buffered, 1 barrier per K-step.
// Reshape/transpose permutations are folded into GEMM epilogues.
// ---------------------------------------------------------------------------

typedef __bf16 bf16;
typedef __bf16 bf16x16 __attribute__((ext_vector_type(16)));
typedef float  f32x8   __attribute__((ext_vector_type(8)));

namespace dcma {

constexpr int  kB   = 256;
constexpr int  kL   = 197;
constexpr int  kD   = 768;
constexpr int  kDK  = 96;
constexpr int  kLP  = 224;   // L padded for K-dims (multiple of 32)
constexpr int  kLPM = 256;   // L padded for row-dims (multiple of 128)
constexpr long kXY  = (long)kB * kL * kD;

union FragAB { uint4 u[2]; bf16x16 v; };

// CDNA5 async global->LDS copy, 16B per lane, tracked by ASYNCcnt.
__device__ __forceinline__ void async_ld_b128(unsigned lds_off, const void* gaddr) {
  asm volatile("global_load_async_to_lds_b128 %0, %1, off"
               :: "v"(lds_off), "v"((unsigned long long)gaddr)
               : "memory");
}
__device__ __forceinline__ void wait_async0() {
  asm volatile("s_wait_asynccnt 0x0" ::: "memory");
}

// ---------------- epilogue functors ----------------

// k = (x@Wc^T + bc); store permuted: kt[b, f%197, f/197] with f = l*96+dk
struct KProjEpi {
  const float* bias;   // combined bias (96)
  bf16*        kt;     // (B, 256, 96)
  __device__ void operator()(int, int r, int c, float v) const {
    if (c >= kDK) return;
    int b = r / kL, l = r - b * kL;
    int f = l * kDK + c;
    int m = f % kL, ch = f / kL;
    kt[(long)b * kLPM * kDK + (long)m * kDK + ch] = (bf16)(v + bias[c]);
  }
};

// v = x@Wv^T + bv; store transposed per batch: vt[b, d, l]
struct VProjEpi {
  const float* bias;   // (768)
  bf16*        vt;     // (B, 768, 224)
  __device__ void operator()(int, int r, int c, float v) const {
    int b = r / kL, l = r - b * kL;
    vt[(long)b * kD * kLP + (long)c * kLP + l] = (bf16)(v + bias[c]);
  }
};

// energy = k1t @ k2t^T, masked to zero outside 197x197; store e and e^T
struct EnergyEpi {
  bf16* e;    // (B, 256, 224)
  bf16* et;   // (B, 256, 224)
  __device__ void operator()(int b, int r, int c, float v) const {
    bf16 val = (bf16)((r < kL && c < kL) ? v : 0.0f);
    if (c < kLP) e [(long)b * kLPM * kLP + (long)r * kLP + c] = val;
    if (r < kLP) et[(long)b * kLPM * kLP + (long)c * kLP + r] = val;
  }
};

// pre-softmax logits: e = energy @ Wl^T + bl (f32 scratch)
struct AttEpi {
  const float* bias;   // (197)
  float*       e;      // (B, 224, 224) f32
  __device__ void operator()(int b, int r, int c, float v) const {
    if (r < kLP && c < kLP) {
      float bb = (c < kL) ? bias[c] : 0.0f;
      e[(long)b * kLP * kLP + (long)r * kLP + c] = v + bb;
    }
  }
};

// out = gamma * (vt @ attT^T) + residual, flat reinterpretation is identity
struct OutEpi {
  const float* gamma;  // 1 element
  const float* resid;  // x or y, (B,L,D) f32
  float*       out;
  __device__ void operator()(int b, int r, int c, float v) const {
    if (c < kL) {
      long f = (long)b * kL * kD + (long)r * kL + c;
      out[f] = gamma[0] * v + resid[f];
    }
  }
};

// ---------------- WMMA GEMM core: C = A @ W^T ----------------
// A: (M,K) row-major bf16, W: (N,K) row-major bf16.
// Block = WM x WN waves (wave32), wave tile = (MF*16) x (NF*16).
// Double-buffered LDS fed by async global->LDS copies.
template<int MF, int NF, int WM, int WN, class Epi>
__global__ __launch_bounds__(WM * WN * 32)
void gemm_bf16(const bf16* __restrict__ A, int lda, long strideA,
               const bf16* __restrict__ W, int ldw, long strideW,
               int K, Epi epi)
{
  constexpr int BM  = WM * MF * 16;
  constexpr int BN  = WN * NF * 16;
  constexpr int NT  = WM * WN * 32;
  constexpr int LS  = 48;                 // LDS row stride (bf16), 96B, 16B aligned
  constexpr int ACH = BM * 4;             // 16B chunks per A K-slice
  constexpr int WCH = BN * 4;
  constexpr int AIT = (ACH + NT - 1) / NT;
  constexpr int WIT = (WCH + NT - 1) / NT;
  constexpr unsigned ABUF = (unsigned)BM * LS * 2;  // bytes per A buffer
  constexpr unsigned WBUF = (unsigned)BN * LS * 2;  // bytes per W buffer
  __shared__ bf16 shA[2][BM * LS];
  __shared__ bf16 shW[2][BN * LS];

  const int tid  = threadIdx.x;
  const int lane = tid & 31;
  const int wave = tid >> 5;
  const int wm   = wave % WM;
  const int wn   = wave / WM;
  const int bz   = blockIdx.z;
  const long tileM = (long)blockIdx.x * BM;
  const long tileN = (long)blockIdx.y * BN;
  A += (long)bz * strideA;
  W += (long)bz * strideW;

  // per-thread staging chunk coordinates: global source row/kc, LDS dest offset
  const bf16* gA[AIT]; const bf16* gW[WIT];
  unsigned    lA[AIT]; unsigned    lW[WIT];
  bool        aOk[AIT], wOk[WIT];
  #pragma unroll
  for (int it = 0; it < AIT; ++it) {
    int ch = tid + it * NT;
    aOk[it] = (ACH % NT == 0) || (ch < ACH);
    int row = ch >> 2, kc = ch & 3;
    gA[it] = A + (tileM + row) * (long)lda + kc * 8;
    lA[it] = (unsigned)(size_t)&shA[0][row * LS + kc * 8];
  }
  #pragma unroll
  for (int it = 0; it < WIT; ++it) {
    int ch = tid + it * NT;
    wOk[it] = (WCH % NT == 0) || (ch < WCH);
    int row = ch >> 2, kc = ch & 3;
    gW[it] = W + (tileN + row) * (long)ldw + kc * 8;
    lW[it] = (unsigned)(size_t)&shW[0][row * LS + kc * 8];
  }

  // ---- prologue: async-stage K-slice 0 into buffer 0 ----
  #pragma unroll
  for (int it = 0; it < AIT; ++it)
    if (aOk[it]) async_ld_b128(lA[it], gA[it]);
  #pragma unroll
  for (int it = 0; it < WIT; ++it)
    if (wOk[it]) async_ld_b128(lW[it], gW[it]);
  wait_async0();
  __syncthreads();

  f32x8 acc[MF][NF];
  #pragma unroll
  for (int i = 0; i < MF; ++i)
    #pragma unroll
    for (int j = 0; j < NF; ++j) acc[i][j] = (f32x8)0.0f;

  const int m16 = lane & 15;
  const int kh  = lane >> 4;
  unsigned cur = 0;

  for (int k0 = 0; k0 < K; k0 += 32) {
    const bool pre = (k0 + 32 < K);
    const unsigned nxt = cur ^ 1u;
    if (pre) {
      // issue async copies of the next K-slice into the other buffer;
      // they complete while the WMMAs below consume the current buffer.
      const int kn = k0 + 32;
      #pragma unroll
      for (int it = 0; it < AIT; ++it)
        if (aOk[it]) async_ld_b128(lA[it] + nxt * ABUF, gA[it] + kn);
      #pragma unroll
      for (int it = 0; it < WIT; ++it)
        if (wOk[it]) async_ld_b128(lW[it] + nxt * WBUF, gW[it] + kn);
    }

    // A frag (16x32, 16-bit): lane half kh holds K chunks {kh*8..+8} and {16+kh*8..+8}
    FragAB fa[MF];
    #pragma unroll
    for (int i = 0; i < MF; ++i) {
      const bf16* p = &shA[cur][(wm * MF * 16 + i * 16 + m16) * LS];
      fa[i].u[0] = *(const uint4*)(p + kh * 8);
      fa[i].u[1] = *(const uint4*)(p + 16 + kh * 8);
    }
    // B frag (32x16): lane n holds column n (= row n of W), K chunks {kh*8..+8} and {16+kh*8..+8}
    FragAB fb[NF];
    #pragma unroll
    for (int j = 0; j < NF; ++j) {
      const bf16* p = &shW[cur][(wn * NF * 16 + j * 16 + m16) * LS];
      fb[j].u[0] = *(const uint4*)(p + kh * 8);
      fb[j].u[1] = *(const uint4*)(p + 16 + kh * 8);
    }
    #pragma unroll
    for (int i = 0; i < MF; ++i)
      #pragma unroll
      for (int j = 0; j < NF; ++j)
        acc[i][j] = __builtin_amdgcn_wmma_f32_16x16x32_bf16(
            false, fa[i].v, false, fb[j].v, (short)0, acc[i][j], false, false);

    if (pre) {
      wait_async0();        // this wave's async copies into 'nxt' are done
      __syncthreads();      // everyone finished reading 'cur' and filling 'nxt'
      cur = nxt;
    }
  }

  // C layout: VGPR e -> row = e + 8*kh, col = lane&15
  #pragma unroll
  for (int i = 0; i < MF; ++i)
    #pragma unroll
    for (int j = 0; j < NF; ++j)
      #pragma unroll
      for (int e = 0; e < 8; ++e) {
        int r = (int)tileM + wm * MF * 16 + i * 16 + kh * 8 + e;
        int c = (int)tileN + wn * NF * 16 + j * 16 + m16;
        epi(bz, r, c, acc[i][j][e]);
      }
}

// ---------------- small prep / softmax kernels ----------------

__global__ void cvt_f32_bf16(const float* __restrict__ in, bf16* __restrict__ out, long n) {
  long i = (long)blockIdx.x * blockDim.x + threadIdx.x;
  if (i < n) out[i] = (bf16)in[i];
}

// Wc[dk,d] = sum_j Wks[dk,j] * Wk[j,d]   (fold two linears into one)
__global__ void combine_w(const float* __restrict__ Wks, const float* __restrict__ Wk,
                          bf16* __restrict__ Wc) {
  int idx = blockIdx.x * 256 + threadIdx.x;
  if (idx >= kDK * kD) return;
  int dk = idx / kD, d = idx - dk * kD;
  float s = 0.0f;
  for (int j = 0; j < kDK; ++j) s += Wks[dk * kDK + j] * Wk[(long)j * kD + d];
  Wc[idx] = (bf16)s;
}

__global__ void combine_b(const float* __restrict__ Wks, const float* __restrict__ bk,
                          const float* __restrict__ bks, float* __restrict__ bc) {
  int dk = threadIdx.x;
  if (dk >= kDK) return;
  float s = bks[dk];
  for (int j = 0; j < kDK; ++j) s += Wks[dk * kDK + j] * bk[j];
  bc[dk] = s;
}

// Wl (197x197 f32) -> (256x224) bf16 zero-padded
__global__ void pad_wl(const float* __restrict__ Wl, bf16* __restrict__ Wlb) {
  int idx = blockIdx.x * 256 + threadIdx.x;
  if (idx >= kLPM * kLP) return;
  int r = idx / kLP, c = idx - r * kLP;
  Wlb[idx] = (bf16)((r < kL && c < kL) ? Wl[r * kL + c] : 0.0f);
}

// softmax over row l of e[b], write transposed into attT[b, m, l] (bf16)
__global__ void softmax_to_T(const float* __restrict__ e, bf16* __restrict__ attT) {
  int l = blockIdx.x;
  int b = blockIdx.y;
  const float* row = e + (long)b * kLP * kLP + (long)l * kLP;
  __shared__ float red[256];
  int t = threadIdx.x;
  float v = (t < kL) ? row[t] : -3.0e38f;
  red[t] = v; __syncthreads();
  for (int s = 128; s > 0; s >>= 1) { if (t < s) red[t] = fmaxf(red[t], red[t + s]); __syncthreads(); }
  float mx = red[0]; __syncthreads();
  float ex = (t < kL) ? __expf(v - mx) : 0.0f;
  red[t] = ex; __syncthreads();
  for (int s = 128; s > 0; s >>= 1) { if (t < s) red[t] += red[t + s]; __syncthreads(); }
  float inv = 1.0f / red[0];
  if (t < kL) attT[(long)b * kLPM * kLP + (long)t * kLP + l] = (bf16)(ex * inv);
}

} // namespace dcma

// ---------------- host launcher ----------------

extern "C" void kernel_launch(void* const* d_in, const int* in_sizes, int n_in,
                              void* d_out, int out_size, void* d_ws, size_t ws_size,
                              hipStream_t stream) {
  using namespace dcma;
  (void)in_sizes; (void)n_in; (void)out_size;

  const float* x    = (const float*)d_in[0];
  const float* y    = (const float*)d_in[1];
  const float* Wk1  = (const float*)d_in[2];
  const float* bk1  = (const float*)d_in[3];
  const float* Wk2  = (const float*)d_in[4];
  const float* bk2  = (const float*)d_in[5];
  const float* Wks  = (const float*)d_in[6];
  const float* bks  = (const float*)d_in[7];
  const float* Wl1  = (const float*)d_in[8];
  const float* bl1  = (const float*)d_in[9];
  const float* Wl2  = (const float*)d_in[10];
  const float* bl2  = (const float*)d_in[11];
  const float* Wv1  = (const float*)d_in[12];
  const float* bv1  = (const float*)d_in[13];
  const float* Wv2  = (const float*)d_in[14];
  const float* bv2  = (const float*)d_in[15];
  const float* g1   = (const float*)d_in[16];
  const float* g2   = (const float*)d_in[17];
  float* out = (float*)d_out;

  // workspace carve-out (256B aligned)
  char* base = (char*)d_ws;
  size_t off = 0;
  auto carve = [&](size_t bytes) -> char* {
    char* p = base + off;
    off = (off + bytes + 255) & ~(size_t)255;
    return p;
  };
  bf16*  xb    = (bf16*)carve(kXY * 2);
  bf16*  yb    = (bf16*)carve(kXY * 2);
  bf16*  Wc1b  = (bf16*)carve((size_t)kDK * kD * 2);
  bf16*  Wc2b  = (bf16*)carve((size_t)kDK * kD * 2);
  bf16*  Wv1b  = (bf16*)carve((size_t)kD * kD * 2);
  bf16*  Wv2b  = (bf16*)carve((size_t)kD * kD * 2);
  bf16*  Wl1b  = (bf16*)carve((size_t)kLPM * kLP * 2);
  bf16*  Wl2b  = (bf16*)carve((size_t)kLPM * kLP * 2);
  float* bc1   = (float*)carve(kDK * 4);
  float* bc2   = (float*)carve(kDK * 4);
  bf16*  k1t   = (bf16*)carve((size_t)kB * kLPM * kDK * 2);
  bf16*  k2t   = (bf16*)carve((size_t)kB * kLPM * kDK * 2);
  bf16*  eb    = (bf16*)carve((size_t)kB * kLPM * kLP * 2);
  bf16*  etb   = (bf16*)carve((size_t)kB * kLPM * kLP * 2);
  float* ework = (float*)carve((size_t)kB * kLP * kLP * 4);
  size_t attBytes = (size_t)kB * kLPM * kLP * 2;
  bf16*  att1t = (bf16*)carve(attBytes);
  bf16*  att2t = (bf16*)carve(attBytes);
  bf16*  v1t   = (bf16*)carve((size_t)kB * kD * kLP * 2);
  bf16*  v2t   = (bf16*)carve((size_t)kB * kD * kLP * 2);
  if (off > ws_size) return;   // insufficient scratch; bail deterministically

  // ---- prep ----
  long nconv = kXY;
  cvt_f32_bf16<<<dim3((unsigned)((nconv + 255) / 256)), 256, 0, stream>>>(x, xb, nconv);
  cvt_f32_bf16<<<dim3((unsigned)((nconv + 255) / 256)), 256, 0, stream>>>(y, yb, nconv);
  long nwv = (long)kD * kD;
  cvt_f32_bf16<<<dim3((unsigned)((nwv + 255) / 256)), 256, 0, stream>>>(Wv1, Wv1b, nwv);
  cvt_f32_bf16<<<dim3((unsigned)((nwv + 255) / 256)), 256, 0, stream>>>(Wv2, Wv2b, nwv);
  combine_w<<<dim3((kDK * kD + 255) / 256), 256, 0, stream>>>(Wks, Wk1, Wc1b);
  combine_w<<<dim3((kDK * kD + 255) / 256), 256, 0, stream>>>(Wks, Wk2, Wc2b);
  combine_b<<<dim3(1), 128, 0, stream>>>(Wks, bk1, bks, bc1);
  combine_b<<<dim3(1), 128, 0, stream>>>(Wks, bk2, bks, bc2);
  pad_wl<<<dim3((kLPM * kLP + 255) / 256), 256, 0, stream>>>(Wl1, Wl1b);
  pad_wl<<<dim3((kLPM * kLP + 255) / 256), 256, 0, stream>>>(Wl2, Wl2b);
  hipMemsetAsync(att1t, 0, attBytes, stream);
  hipMemsetAsync(att2t, 0, attBytes, stream);

  const long sE = (long)kLPM * kLP;   // per-batch stride of energy/att buffers
  const long sK = (long)kLPM * kDK;   // per-batch stride of k1t/k2t
  const long sV = (long)kD * kLP;     // per-batch stride of v1t/v2t

  // ---- k projections (M = B*L = 50432 = 394*128, N = 96, K = 768) ----
  gemm_bf16<2, 3, 4, 2, KProjEpi><<<dim3(394, 1, 1), 256, 0, stream>>>(
      xb, kD, 0, Wc1b, kD, 0, kD, KProjEpi{bc1, k1t});
  gemm_bf16<2, 3, 4, 2, KProjEpi><<<dim3(394, 1, 1), 256, 0, stream>>>(
      yb, kD, 0, Wc2b, kD, 0, kD, KProjEpi{bc2, k2t});

  // ---- energy = k1t @ k2t^T per batch (197x197, K=96) ----
  gemm_bf16<2, 4, 4, 2, EnergyEpi><<<dim3(2, 2, kB), 256, 0, stream>>>(
      k1t, kDK, sK, k2t, kDK, sK, kDK, EnergyEpi{eb, etb});

  // ---- att1 = softmax(energy @ Wl1^T + bl1) -> att1t (transposed) ----
  gemm_bf16<2, 4, 4, 2, AttEpi><<<dim3(2, 2, kB), 256, 0, stream>>>(
      eb, kLP, sE, Wl1b, kLP, 0, kLP, AttEpi{bl1, ework});
  softmax_to_T<<<dim3(kL, kB), 256, 0, stream>>>(ework, att1t);

  // ---- att2 = softmax(energy^T @ Wl2^T + bl2) -> att2t ----
  gemm_bf16<2, 4, 4, 2, AttEpi><<<dim3(2, 2, kB), 256, 0, stream>>>(
      etb, kLP, sE, Wl2b, kLP, 0, kLP, AttEpi{bl2, ework});
  softmax_to_T<<<dim3(kL, kB), 256, 0, stream>>>(ework, att2t);

  // ---- v projections (M = 50432, N = 768, K = 768), stored transposed ----
  gemm_bf16<2, 4, 4, 2, VProjEpi><<<dim3(394, 6, 1), 256, 0, stream>>>(
      yb, kD, 0, Wv2b, kD, 0, kD, VProjEpi{bv2, v2t});
  gemm_bf16<2, 4, 4, 2, VProjEpi><<<dim3(394, 6, 1), 256, 0, stream>>>(
      xb, kD, 0, Wv1b, kD, 0, kD, VProjEpi{bv1, v1t});

  // ---- outputs: out_x = gamma1 * (v2t @ att1t^T) + x ; out_y likewise ----
  gemm_bf16<2, 4, 4, 2, OutEpi><<<dim3(6, 2, kB), 256, 0, stream>>>(
      v2t, kLP, sV, att1t, kLP, sE, kLP, OutEpi{g1, x, out});
  gemm_bf16<2, 4, 4, 2, OutEpi><<<dim3(6, 2, kB), 256, 0, stream>>>(
      v1t, kLP, sV, att2t, kLP, sE, kLP, OutEpi{g2, y, out + kXY});
}
